// ClassifierJax_75024488726882
// MI455X (gfx1250) — matrix operand order repaired and checked by reading, not compiled, not run on hardware
//
#include <hip/hip_runtime.h>
#include <math.h>

typedef __attribute__((ext_vector_type(16))) _Float16 v16h;
typedef __attribute__((ext_vector_type(8)))  float    v8f;

#define T_LEN 128
#define HID   128
#define INC   32
#define NB    16      // batch tile per workgroup == WMMA N
#define OUTC  10
#define NW    8       // waves per block (256 threads, wave32)
#define NSTEPS 64
#define K_RTOL 1e-3f
#define K_ATOL 1e-3f
#define K_SAFETY 0.9f
#define K_DT0 0.01f

// Dormand-Prince tableau
__constant__ float c_A[7][6] = {
  {0.f,0.f,0.f,0.f,0.f,0.f},
  {0.2f,0.f,0.f,0.f,0.f,0.f},
  {3.f/40.f, 9.f/40.f, 0.f,0.f,0.f,0.f},
  {44.f/45.f, -56.f/15.f, 32.f/9.f, 0.f,0.f,0.f},
  {19372.f/6561.f, -25360.f/2187.f, 64448.f/6561.f, -212.f/729.f, 0.f,0.f},
  {9017.f/3168.f, -355.f/33.f, 46732.f/5247.f, 49.f/176.f, -5103.f/18656.f, 0.f},
  {35.f/384.f, 0.f, 500.f/1113.f, 125.f/192.f, -2187.f/6784.f, 11.f/84.f} // b row -> y_new
};
__constant__ float c_C[7] = {0.f, 0.2f, 0.3f, 0.8f, 8.f/9.f, 1.f, 1.f};
__constant__ float c_E[7] = {71.f/57600.f, 0.f, -71.f/16695.f, 71.f/1920.f,
                             -17253.f/339200.f, 22.f/525.f, -1.f/40.f};

// Fast transcendental helpers: 1e-3 tolerance + f16 WMMA inputs make 1-ulp
// hardware approximations (v_exp_f32 / v_rcp_f32 / v_tanh_f32) plenty.
__device__ __forceinline__ float fast_tanh(float x) {
#if __has_builtin(__builtin_amdgcn_tanhf)
  return __builtin_amdgcn_tanhf(x);          // gfx1250 v_tanh_f32
#else
  // tanh(x) = 1 - 2/(e^{2x}+1); saturates correctly at +-inf of exp2
  float e = __builtin_amdgcn_exp2f(x * 2.8853900817779268f); // 2*log2(e)
  return 1.f - 2.f * __builtin_amdgcn_rcpf(e + 1.f);
#endif
}
__device__ __forceinline__ float fast_rcp(float x) {
  return __builtin_amdgcn_rcpf(x);
}

// ---------------------------------------------------------------------------
// Repack W1 [128x128] and W2 [4096x128] (row-major f32) into f16 WMMA
// A-fragment order (ISA 7.12.2, 16-bit A 16x32):
//   lane l (0..31): M = l%16; lane-half h=l/16 selects K offset;
//   VGPR v element pair p: K = v<4 ? 2v+p+8h : 16+2(v-4)+p+8h.
// Fragment f = mt*4+kt stored contiguously: dst[(f*32+lane)*16 + e].
// ---------------------------------------------------------------------------
__global__ void __launch_bounds__(256)
repack_kernel(const float* __restrict__ W1, const float* __restrict__ W2,
              _Float16* __restrict__ w1f, _Float16* __restrict__ w2f) {
  int t = blockIdx.x * 256 + threadIdx.x;
  const int W2FL = 256 * 4 * 32;   // frag-lanes for W2 (256 M-tiles x 4 K-tiles)
  const int W1FL = 8 * 4 * 32;     // frag-lanes for W1 (8 M-tiles x 4 K-tiles)
  const float* W;
  _Float16* dst;
  int mt, kt, lane;
  if (t < W2FL) {
    lane = t & 31; int frag = t >> 5; kt = frag & 3; mt = frag >> 2;
    W = W2; dst = w2f + (size_t)t * 16;
  } else if (t < W2FL + W1FL) {
    int t1 = t - W2FL;
    lane = t1 & 31; int frag = t1 >> 5; kt = frag & 3; mt = frag >> 2;
    W = W1; dst = w1f + (size_t)t1 * 16;
  } else {
    return;
  }
  int hlf = lane >> 4, M = lane & 15;
  #pragma unroll
  for (int e = 0; e < 16; ++e) {
    int v = e >> 1, p = e & 1;
    int K = (v < 4) ? (2 * v + p + 8 * hlf) : (16 + 2 * (v - 4) + p + 8 * hlf);
    dst[e] = (_Float16)W[(size_t)(mt * 16 + M) * 128 + kt * 32 + K];
  }
}

// ---------------------------------------------------------------------------
// Persistent RK45 CDE solver. One workgroup = 16 batch elements.
// ---------------------------------------------------------------------------
__global__ void __launch_bounds__(256)
cde_solver(const float* __restrict__ ts, const float* __restrict__ xs,
           const float* __restrict__ b1g, const float* __restrict__ b2g,
           const float* __restrict__ lin_w, const float* __restrict__ lin_b,
           const _Float16* __restrict__ w1f, const _Float16* __restrict__ w2f,
           float* __restrict__ out) {
  __shared__ float kb[6][NB * HID];                 // k1..k6 (k7 aliases kb[1])
  __shared__ float yl[NB * HID];
  __shared__ float ynl[NB * HID];
  __shared__ __align__(16) _Float16 yfragl[4][32][16];   // B-frags of y_stage
  __shared__ __align__(16) _Float16 h1fragl[4][32][16];  // B-frags of relu(h1)
  __shared__ float dXl[NB][INC];
  __shared__ float tsl[T_LEN];
  __shared__ float tl[NB], dtl[NB], dtcl[NB], errs[NB], accl[NB];

  const int tid  = threadIdx.x;
  const int lane = tid & 31;
  const int w    = tid >> 5;
  const int bbase = blockIdx.x * NB;

  // ---- init ----
  if (tid < T_LEN) tsl[tid] = ts[tid];
  if (tid < NB) { tl[tid] = ts[0]; dtl[tid] = K_DT0; }
  for (int e = 0; e < (NB * HID) / 256; ++e) yl[tid + e * 256] = 0.f;
  __syncthreads();
  const float t_end = tsl[T_LEN - 1];

  #pragma unroll 1
  for (int step = 0; step < NSTEPS; ++step) {
    if (tid < NB) { dtcl[tid] = fminf(dtl[tid], t_end - tl[tid]); errs[tid] = 0.f; }
    __syncthreads();

    #pragma unroll 1
    for (int s = 0; s < 7; ++s) {
      // k7 (FSAL stage) writes into kb[1]; k2 is dead after stage 6 and
      // has zero coefficient in both the b-row and the error estimate.
      float* kout = kb[(s == 6) ? 1 : s];

      // (a) y_stage = y + dt_c * sum_j A[s][j]*k_j  ->  f16 B-fragments
      for (int e = 0; e < (NB * HID) / 256; ++e) {
        int idx = tid + e * 256;
        int n = idx >> 7, hh = idx & 127;
        float acc = yl[idx];
        float dtc = dtcl[n];
        for (int j = 0; j < 6; ++j) {
          float a = c_A[s][j];
          if (a != 0.f) acc += dtc * a * kb[j][idx];
        }
        if (s == 6) ynl[idx] = acc;   // stage 7 evaluates at y_new (FSAL)
        int kt = hh >> 5, rr = hh & 31;
        yfragl[kt][n + (((rr >> 3) & 1) << 4)][(rr & 7) + ((rr >> 4) << 3)] = (_Float16)acc;
        kout[idx] = 0.f;              // zero accumulation target for k_{s+1}
      }

      // (a2) dXdt(t_s) via Hermite derivative on the grid
      {
        int n = tid >> 4, cc = tid & 15;
        float t_s = tl[n] + c_C[s] * dtcl[n];
        int lo = 0, hi = T_LEN - 1;
        while (hi - lo > 1) {
          int mid = (lo + hi) >> 1;
          if (tsl[mid] <= t_s) lo = mid; else hi = mid;
        }
        int i = (lo > T_LEN - 2) ? (T_LEN - 2) : lo;
        float h   = tsl[i + 1] - tsl[i];
        float rh  = fast_rcp(h);
        float sfr = (t_s - tsl[i]) * rh;
        float dh00 = 6.f * sfr * sfr - 6.f * sfr;
        float dh10 = 3.f * sfr * sfr - 4.f * sfr + 1.f;
        float dh11 = 3.f * sfr * sfr - 2.f * sfr;
        const float* xb = xs + (size_t)(bbase + n) * T_LEN * INC;
        #pragma unroll
        for (int cq = 0; cq < 2; ++cq) {
          int c = cc + cq * 16;
          float xi  = xb[i * INC + c];
          float xip = xb[(i + 1) * INC + c];
          float mi  = (i == 0)
                    ? (xb[INC + c] - xb[c]) * fast_rcp(tsl[1] - tsl[0])
                    : (xi - xb[(i - 1) * INC + c]) * fast_rcp(tsl[i] - tsl[i - 1]);
          float mip = (xip - xi) * rh;
          dXl[n][c] = dh00 * (xi - xip) * rh + dh10 * mi + dh11 * mip;
        }
      }
      __syncthreads();

      // (b) GEMM1: H1 = relu(W1 @ Ystage + b1); repack into B-fragments
      {
        v8f acc; for (int i = 0; i < 8; ++i) acc[i] = 0.f;
        #pragma unroll
        for (int kt = 0; kt < 4; ++kt) {
          v16h a = *(const v16h*)(w1f + (size_t)(((w << 2) + kt) * 32 + lane) * 16);
          v16h b = *(const v16h*)(&yfragl[kt][lane][0]);
          acc = __builtin_amdgcn_wmma_f32_16x16x32_f16(false, a, false, b,
                                                       (short)0, acc, false, false);
        }
        int hlf = lane >> 4, col = lane & 15;
        #pragma unroll
        for (int v = 0; v < 8; ++v) {
          int row = (w << 4) + v + (hlf << 3);
          float val = fmaxf(acc[v] + b1g[row], 0.f);
          int kt2 = row >> 5, rr = row & 31;
          h1fragl[kt2][col + (((rr >> 3) & 1) << 4)][(rr & 7) + ((rr >> 4) << 3)] = (_Float16)val;
        }
      }
      __syncthreads();

      // (c) GEMM2: tanh(W2 @ H1 + b2) fused with dX-weighted reduction -> k_s
      // Explicit ping-pong software pipeline: fragments for tile mt+8 are in
      // flight while the WMMAs + epilogue of tile mt execute; no reg rotation.
      {
        const int hlf = lane >> 4, n = lane & 15;

        auto load_af = [&](int mt, v16h (&af)[4]) {
          #pragma unroll
          for (int kt = 0; kt < 4; ++kt)
            af[kt] = *(const v16h*)(w2f +
                       (size_t)(((mt << 2) + kt) * 32 + lane) * 16);
        };
        auto do_tile = [&](int mt, const v16h (&af)[4]) {
          v8f acc; for (int i = 0; i < 8; ++i) acc[i] = 0.f;
          #pragma unroll
          for (int kt = 0; kt < 4; ++kt) {
            v16h b = *(const v16h*)(&h1fragl[kt][lane][0]);
            acc = __builtin_amdgcn_wmma_f32_16x16x32_f16(false, af[kt], false, b,
                                                         (short)0, acc, false, false);
          }
          float partial = 0.f;
          #pragma unroll
          for (int v = 0; v < 8; ++v) {
            int row = (mt << 4) + v + (hlf << 3);   // row = h*32 + c
            float f = fast_tanh(acc[v] + b2g[row]);
            partial += f * dXl[n][row & 31];
          }
          partial += __shfl_xor(partial, 16, 32);   // fold M rows across lane halves
          if (lane < 16) atomicAdd(&kout[n * HID + (mt >> 1)], partial);
        };

        v16h A0[4], A1[4];
        load_af(w, A0);
        #pragma unroll 1
        for (int it = 0; it < 16; ++it) {           // 32 tiles, 2 per iteration
          int mt = w + it * 2 * NW;
          load_af((mt + NW) & 255, A1);             // wrap on last pair: unused
          do_tile(mt, A0);
          load_af((mt + 2 * NW) & 255, A0);
          do_tile(mt + NW, A1);
        }
      }
      __syncthreads();
    } // stages

    // ---- embedded error + step control ----
    for (int e = 0; e < (NB * HID) / 256; ++e) {
      int idx = tid + e * 256;
      int n = idx >> 7;
      float ev = dtcl[n] * (c_E[0] * kb[0][idx] + c_E[2] * kb[2][idx] +
                            c_E[3] * kb[3][idx] + c_E[4] * kb[4][idx] +
                            c_E[5] * kb[5][idx] + c_E[6] * kb[1][idx]); // k7 in kb[1]
      float sc = K_ATOL + K_RTOL * fmaxf(fabsf(yl[idx]), fabsf(ynl[idx]));
      float r = ev * fast_rcp(sc);
      atomicAdd(&errs[n], r * r);
    }
    __syncthreads();
    if (tid < NB) {
      float err = sqrtf(errs[tid] / (float)HID);
      bool accept = err <= 1.f;
      // err^-0.2 via exp2/log2 (err clamped > 0)
      float ec = fmaxf(err, 1e-9f);
      float factor = K_SAFETY *
          __builtin_amdgcn_exp2f(-0.2f * __builtin_amdgcn_logf(ec));
      factor = fminf(fmaxf(factor, 0.2f), 10.f);
      bool done = tl[tid] >= t_end - 1e-8f;
      bool upd = accept && !done;
      if (upd) tl[tid] += dtcl[tid];
      if (!done) dtl[tid] = dtcl[tid] * factor;
      accl[tid] = upd ? 1.f : 0.f;
    }
    __syncthreads();
    for (int e = 0; e < (NB * HID) / 256; ++e) {
      int idx = tid + e * 256;
      if (accl[idx >> 7] != 0.f) yl[idx] = ynl[idx];
    }
    __syncthreads();
  } // steps

  // ---- final linear: out = yT @ lin_w.T + lin_b ----
  if (tid < NB * OUTC) {
    int n = tid / OUTC, o = tid % OUTC;
    float acc = lin_b[o];
    #pragma unroll 4
    for (int h = 0; h < HID; ++h) acc += yl[n * HID + h] * lin_w[o * HID + h];
    out[(bbase + n) * OUTC + o] = acc;
  }
}

extern "C" void kernel_launch(void* const* d_in, const int* in_sizes, int n_in,
                              void* d_out, int out_size, void* d_ws, size_t ws_size,
                              hipStream_t stream) {
  const float* ts    = (const float*)d_in[0];
  const float* xs    = (const float*)d_in[1];
  const float* W1    = (const float*)d_in[2];
  const float* b1    = (const float*)d_in[3];
  const float* W2    = (const float*)d_in[4];
  const float* b2    = (const float*)d_in[5];
  const float* lin_w = (const float*)d_in[6];
  const float* lin_b = (const float*)d_in[7];

  _Float16* w2f = (_Float16*)d_ws;                 // 256*4*32*16 halves = 1 MB
  _Float16* w1f = w2f + (size_t)256 * 4 * 32 * 16; // + 32 KB

  // 32768 W2 frag-lanes + 1024 W1 frag-lanes = 33792 threads
  repack_kernel<<<132, 256, 0, stream>>>(W1, W2, w1f, w2f);
  cde_solver<<<8, 256, 0, stream>>>(ts, xs, b1, b2, lin_w, lin_b, w1f, w2f,
                                    (float*)d_out);
}
